// RPEAttention_40492951666722
// MI455X (gfx1250) — matrix operand order repaired and checked by reading, not compiled, not run on hardware
//
#include <hip/hip_runtime.h>

typedef __bf16 bf16;
typedef __bf16 v16bf __attribute__((ext_vector_type(16)));
typedef float  v8f   __attribute__((ext_vector_type(8)));

constexpr int B_ = 2, D_ = 1024, T_ = 32, C_ = 512, H_ = 8, F_ = 64, TE_ = 512;
constexpr long MROWS = (long)B_ * D_ * T_;   // 65536 token rows

// LDS staging tile: 32 rows x 64 bf16, padded to 72-element (144B) pitch so
// async b128 chunks stay 16B aligned and banks are spread.
constexpr int STG_PITCH = 72;                 // elements
constexpr int STG_BYTES = 32 * STG_PITCH * 2; // 4608 B per tile

union BP { unsigned u; bf16 h[2]; };

// -------- CDNA5 async global->LDS copy (ASYNCcnt-tracked) --------
// lds_ptr must be a real pointer into __shared__ memory: low 32 bits of the
// flat address are the LDS byte offset (VDST operand), and the ptrtoint makes
// the shared array escape into the asm so its readers are not folded to undef.
#define ASYNC_LOAD_B128(lds_ptr, gaddr)                                       \
    asm volatile("global_load_async_to_lds_b128 %0, %1, off"                  \
                 :: "v"((unsigned)(unsigned long long)(lds_ptr)),             \
                    "v"((unsigned long long)(gaddr)) : "memory")
#define WAIT_ASYNC() asm volatile("s_wait_asynccnt 0x0" ::: "memory")

static __device__ inline v8f vzero() {
    v8f z;
#pragma unroll
    for (int i = 0; i < 8; ++i) z[i] = 0.f;
    return z;
}

static __device__ inline v8f wmma_bf16(v16bf a, v16bf b, v8f c) {
    return __builtin_amdgcn_wmma_f32_16x16x32_bf16(false, a, false, b, (short)0, c, false, false);
}

// A-matrix fragment (16x32 bf16): row = per-lane row pointer (M = lane&15),
// kbase = kc + 8*(lane>>4). Contiguous K -> dword-pair loads.
static __device__ inline v16bf load_a_cont(const bf16* row, int kbase) {
    v16bf a;
#pragma unroll
    for (int i = 0; i < 8; ++i) {
        int K0 = kbase + ((i >> 2) << 4) + ((i & 3) << 1);
        BP bp; bp.u = *reinterpret_cast<const unsigned*>(row + K0);
        a[2 * i] = bp.h[0]; a[2 * i + 1] = bp.h[1];
    }
    return a;
}

// A fragment from an f32 LDS row (softmax probabilities -> bf16)
static __device__ inline v16bf load_a_lds(const float* row, int kbase) {
    v16bf a;
#pragma unroll
    for (int e = 0; e < 16; ++e) {
        int i = e >> 1;
        int K = kbase + ((i >> 2) << 4) + ((i & 3) << 1) + (e & 1);
        a[e] = (bf16)row[K];
    }
    return a;
}

// B-matrix fragment (32x16 bf16): N = n0 + (lane&15), element e holds
// K = 8*(lane>>4) + (e&7) + 16*(e>>3).
static __device__ inline v16bf load_b_cont(const bf16* base, int strideK, int n0) {
    int l = threadIdx.x & 31;
    const bf16* p = base + n0 + (l & 15);
    v16bf b;
#pragma unroll
    for (int e = 0; e < 16; ++e) {
        int K = ((l >> 4) << 3) + (e & 7) + ((e >> 3) << 4);
        b[e] = p[(size_t)K * strideK];
    }
    return b;
}

// B fragment from LDS tile laid out [K=32][N=64] with STG_PITCH (GEMM path)
static __device__ inline v16bf load_b_ldsKN(const bf16* tile, int n0) {
    int l = threadIdx.x & 31;
    const bf16* p = tile + n0 + (l & 15);
    v16bf b;
#pragma unroll
    for (int e = 0; e < 16; ++e) {
        int K = ((l >> 4) << 3) + (e & 7) + ((e >> 3) << 4);
        b[e] = p[(size_t)K * STG_PITCH];
    }
    return b;
}

// B fragment from LDS tile laid out [N=32][K=64] (attention implicit-transpose path)
static __device__ inline v16bf load_b_ldsNK(const bf16* tile, int kc, int n0) {
    int l = threadIdx.x & 31;
    v16bf b;
#pragma unroll
    for (int e = 0; e < 16; ++e) b[e] = tile[(size_t)(n0 + (l & 15)) * STG_PITCH + kc + ((l >> 4) << 3) + (e & 7) + ((e >> 3) << 4)];
    return b;
}

// wave-level async stage of a 32x64 bf16 tile (row stride in elements) into an
// LDS tile. Each lane copies one 128B row as 8 async b128 chunks; wave waits
// on ASYNCcnt (wave-local, no workgroup barrier needed for private tiles).
static __device__ inline void stage_tile_wave(const bf16* gbase, size_t rowStrideElems,
                                              bf16* ldsTile) {
    int lane = threadIdx.x & 31;
    const char* grow = (const char*)(gbase + rowStrideElems * lane);
    char* lrow = (char*)(ldsTile + (size_t)lane * STG_PITCH);
#pragma unroll
    for (int c = 0; c < 8; ++c) ASYNC_LOAD_B128(lrow + c * 16, grow + c * 16);
    WAIT_ASYNC();
}

// ---------------- LayerNorm -> bf16 ----------------
__global__ void __launch_bounds__(256) ln_kernel(const float* __restrict__ x,
                                                 const float* __restrict__ w,
                                                 const float* __restrict__ b,
                                                 bf16* __restrict__ xn) {
    int wid = threadIdx.x >> 5, lane = threadIdx.x & 31;
    long row = (long)blockIdx.x * 8 + wid;
    const float* xr = x + row * C_;
    float s = 0.f, s2 = 0.f;
    for (int i = lane; i < C_; i += 32) { float t = xr[i]; s += t; s2 += t * t; }
#pragma unroll
    for (int m = 16; m >= 1; m >>= 1) { s += __shfl_xor(s, m, 32); s2 += __shfl_xor(s2, m, 32); }
    float mu = s / C_;
    float var = s2 / C_ - mu * mu;
    float r = rsqrtf(var + 1e-5f);
    bf16* o = xn + row * C_;
    for (int i = lane; i < C_; i += 32) o[i] = (bf16)((xr[i] - mu) * r * w[i] + b[i]);
}

// ---------------- weight transpose + f32->bf16 :  Wt[k*N+n] = W[n*K+k] ----------------
__global__ void tcvt_kernel(const float* __restrict__ src, bf16* __restrict__ dst,
                            int N, int K) {
    long idx = (long)blockIdx.x * 256 + threadIdx.x;
    if (idx >= (long)N * K) return;
    int n = (int)(idx / K), k = (int)(idx % K);
    dst[(size_t)k * N + n] = (bf16)src[idx];
}

__global__ void cvt_kernel(const float* __restrict__ s, bf16* __restrict__ d, long n) {
    long i = (long)blockIdx.x * 256 + threadIdx.x;
    if (i < n) d[i] = (bf16)s[i];
}

// -------- generic WMMA GEMM: Y[M,N] = A[M,K] * Wt[K,N] (+bias)(+resid) --------
// B tile (32x64) staged in LDS via async-to-LDS, double-buffered: the copy of
// tile i+1 overlaps the 4 WMMAs of tile i; all 4 waves share the staged tile.
__global__ void __launch_bounds__(128) gemm_kernel(const bf16* __restrict__ A,
                                                   const bf16* __restrict__ Wt,
                                                   const float* __restrict__ bias,
                                                   const float* __restrict__ resid,
                                                   float* __restrict__ outf,
                                                   bf16* __restrict__ outh,
                                                   int M, int N, int K) {
    __shared__ bf16 s_bt[2][32 * STG_PITCH];   // 2 x 4608 B

    int wave = threadIdx.x >> 5, lane = threadIdx.x & 31;
    int m0 = blockIdx.x * 64 + wave * 16;
    int n0 = blockIdx.y * 64;
    int Ml = lane & 15, g = lane >> 4, kb8 = g << 3;
    const bf16* arow = A + (size_t)(m0 + Ml) * K;

    // async-issue one B tile: 256 16B chunks, 2 per thread (a quarter of a 128B row per thread)
    auto issue = [&](int buf, int kc) {
#pragma unroll
        for (int j = 0; j < 2; ++j) {
            int ch = threadIdx.x * 2 + j;      // 0..255
            int row = ch >> 3;                 // k-row 0..31
            int col = ch & 7;                  // 16B chunk within 128B row
            char* lp = (char*)&s_bt[buf][(size_t)row * STG_PITCH] + col * 16;
            const char* gp = (const char*)(Wt + (size_t)(kc + row) * N + n0) + col * 16;
            ASYNC_LOAD_B128(lp, gp);
        }
    };

    v8f acc[4];
#pragma unroll
    for (int t = 0; t < 4; ++t) acc[t] = vzero();

    int nk = K / 32;
    issue(0, 0);
    for (int it = 0; it < nk; ++it) {
        int kc = it * 32;
        WAIT_ASYNC();
        __syncthreads();                        // staged tile complete, prev readers done
        if (it + 1 < nk) issue((it + 1) & 1, kc + 32);
        __builtin_prefetch(arow + kc + 128, 0, 3);   // global_prefetch_b8 on A stream
        v16bf a = load_a_cont(arow, kc + kb8);
        const bf16* tile = &s_bt[it & 1][0];
#pragma unroll
        for (int t = 0; t < 4; ++t) {
            v16bf b = load_b_ldsKN(tile, t * 16);
            acc[t] = wmma_bf16(a, b, acc[t]);
        }
    }
#pragma unroll
    for (int t = 0; t < 4; ++t) {
#pragma unroll
        for (int r = 0; r < 8; ++r) {
            int m = m0 + r + kb8;
            int n = n0 + t * 16 + Ml;
            float v = acc[t][r];
            if (bias) v += bias[n];
            if (resid) v += resid[(size_t)m * N + n];
            if (outf) outf[(size_t)m * N + n] = v;
            else      outh[(size_t)m * N + n] = (bf16)v;
        }
    }
}

// ---------------- RPE embedding + SiLU -> bf16 GEMM input ----------------
__global__ void rpe_silu_kernel(const float* __restrict__ te, const float* __restrict__ dw,
                                const float* __restrict__ db, const int* __restrict__ fi,
                                bf16* __restrict__ embA) {
    long idx = (long)blockIdx.x * 256 + threadIdx.x;   // over B*T*T*C
    if (idx >= (long)B_ * T_ * T_ * C_) return;
    int c = (int)(idx % C_);
    long r = idx / C_;
    int s = (int)(r % T_);
    long r2 = r / T_;
    int t = (int)(r2 % T_);
    int b = (int)(r2 / T_);
    float rd = (float)(fi[b * T_ + t] - fi[b * T_ + s]);
    float d0 = log1pf(fmaxf(rd, 0.f));
    float d1 = log1pf(fmaxf(-rd, 0.f));
    float d2 = (rd == 0.f) ? 1.f : 0.f;
    float e = te[((long)b * T_ + t) * C_ + c]
            + d0 * dw[c * 3 + 0] + d1 * dw[c * 3 + 1] + d2 * dw[c * 3 + 2] + db[c];
    float si = e / (1.f + __expf(-e));
    embA[idx] = (bf16)si;
}

// ---------------- fused RPE attention core ----------------
// block = (b, h, 16-d tile); 4 waves. Dynamic LDS: 64KB f32 logit tile +
// 4 wave-private staging tiles filled with async-to-LDS (ASYNCcnt, wave-local).
__global__ void __launch_bounds__(128) attn_kernel(const bf16* __restrict__ q,
                                                   const bf16* __restrict__ k,
                                                   const bf16* __restrict__ v,
                                                   const bf16* __restrict__ Rq,
                                                   const bf16* __restrict__ Rk,
                                                   const bf16* __restrict__ Rv,
                                                   float* __restrict__ gacc,
                                                   bf16* __restrict__ aout) {
    extern __shared__ char smem[];
    float* s_attn = (float*)smem;                       // [16][32][32] f32 = 64KB
#define SA(dd, t, s) s_attn[(((dd) << 10) + ((t) << 5) + (s))]
    const unsigned stgBase0 = 16 * 32 * 32 * 4;         // 65536

    int wave = threadIdx.x >> 5, lane = threadIdx.x & 31;
    int Ml = lane & 15, g = lane >> 4, kb8 = g << 3;
    bf16* myStg = (bf16*)(smem + stgBase0 + (size_t)wave * STG_BYTES);

    int nblk = D_ / 16;
    int bx = blockIdx.x;
    int d0 = (bx % nblk) * 16;
    int h  = (bx / nblk) % H_;
    int b  = bx / (nblk * H_);
    size_t baseB = (size_t)b * D_ * T_ * C_;
    int hc = h * F_;
    const bf16* Rqb = Rq + (size_t)b * T_ * T_ * C_ + hc;
    const bf16* Rkb = Rk + (size_t)b * T_ * T_ * C_ + hc;
    const bf16* Rvb = Rv + (size_t)b * T_ * T_ * C_ + hc;

    // ---- phase 1: S = q . k^T  (per-d, d-split across waves) ----
    for (int di = 0; di < 4; ++di) {
        int d = d0 + wave * 4 + di, dd = wave * 4 + di;
        const bf16* qd = q + baseB + (size_t)d * T_ * C_ + hc;
        const bf16* kd = k + baseB + (size_t)d * T_ * C_ + hc;
        stage_tile_wave(kd, C_, myStg);                 // LDS tile: [s=32][f=64]
        for (int t0 = 0; t0 < 32; t0 += 16)
            for (int s0 = 0; s0 < 32; s0 += 16) {
                v8f acc = vzero();
                for (int kc = 0; kc < 64; kc += 32) {
                    v16bf a = load_a_cont(qd + (size_t)(t0 + Ml) * C_, kc + kb8);
                    v16bf bb = load_b_ldsNK(myStg, kc, s0);  // B[f][s] = k[s][f]
                    acc = wmma_bf16(a, bb, acc);
                }
#pragma unroll
                for (int r = 0; r < 8; ++r) SA(dd, t0 + r + kb8, s0 + Ml) = acc[r];
            }
    }
    __syncthreads();

    // ---- phase 2: += q . Rk^T  (M = d rows, per-t, t-split) ----
    for (int ti = 0; ti < 8; ++ti) {
        int t = wave + ti * 4;
        stage_tile_wave(Rkb + (size_t)t * T_ * C_, C_, myStg);   // [s][f]
        for (int s0 = 0; s0 < 32; s0 += 16) {
            v8f acc = vzero();
            for (int kc = 0; kc < 64; kc += 32) {
                v16bf a = load_a_cont(q + baseB + (size_t)(d0 + Ml) * T_ * C_ +
                                      (size_t)t * C_ + hc, kc + kb8);
                v16bf bb = load_b_ldsNK(myStg, kc, s0);
                acc = wmma_bf16(a, bb, acc);
            }
#pragma unroll
            for (int r = 0; r < 8; ++r) SA(r + kb8, t, s0 + Ml) += acc[r];
        }
    }
    __syncthreads();

    // ---- phase 3: += k . Rq^T (transposed term; per-s, s-split) ----
    for (int si = 0; si < 8; ++si) {
        int s = wave + si * 4;
        stage_tile_wave(Rqb + (size_t)s * T_ * C_, C_, myStg);   // [t][f]
        for (int t0 = 0; t0 < 32; t0 += 16) {
            v8f acc = vzero();
            for (int kc = 0; kc < 64; kc += 32) {
                v16bf a = load_a_cont(k + baseB + (size_t)(d0 + Ml) * T_ * C_ +
                                      (size_t)s * C_ + hc, kc + kb8);
                v16bf bb = load_b_ldsNK(myStg, kc, t0);
                acc = wmma_bf16(a, bb, acc);
            }
#pragma unroll
            for (int r = 0; r < 8; ++r) SA(r + kb8, t0 + Ml, s) += acc[r];
        }
    }
    __syncthreads();

    // ---- softmax over s (scale 1/8) ----
    for (int i = 0; i < 4; ++i) {
        int rr = threadIdx.x + i * 128;        // 512 rows
        int dd = rr >> 5, t = rr & 31;
        float mx = -1e30f;
        for (int s2 = 0; s2 < 32; ++s2) {
            float vv = SA(dd, t, s2) * 0.125f;
            SA(dd, t, s2) = vv;
            mx = fmaxf(mx, vv);
        }
        float sum = 0.f;
        for (int s2 = 0; s2 < 32; ++s2) {
            float e = __expf(SA(dd, t, s2) - mx);
            SA(dd, t, s2) = e;
            sum += e;
        }
        float inv = 1.f / sum;
        for (int s2 = 0; s2 < 32; ++s2) SA(dd, t, s2) *= inv;
    }
    __syncthreads();

    // ---- phase 4: out = P . v (per-d, d-split) -> global f32 scratch ----
    for (int di = 0; di < 4; ++di) {
        int d = d0 + wave * 4 + di, dd = wave * 4 + di;
        const bf16* vd = v + baseB + (size_t)d * T_ * C_ + hc;
        float* gd = gacc + baseB + (size_t)d * T_ * C_ + hc;
        for (int t0 = 0; t0 < 32; t0 += 16) {
            v16bf a = load_a_lds(&SA(dd, t0 + Ml, 0), kb8);
#pragma unroll
            for (int f0 = 0; f0 < 64; f0 += 16) {
                v16bf bb = load_b_cont(vd, C_, f0);   // B[s][f] = v, natural layout
                v8f acc = wmma_bf16(a, bb, vzero());
#pragma unroll
                for (int r = 0; r < 8; ++r)
                    gd[(size_t)(t0 + r + kb8) * C_ + f0 + Ml] = acc[r];
            }
        }
    }
    __threadfence();
    __syncthreads();

    // ---- phase 5: out += P . Rv (M = d rows, per-t, t-split) ----
    for (int ti = 0; ti < 8; ++ti) {
        int t = wave + ti * 4;
        v16bf a = load_a_lds(&SA(Ml, t, 0), kb8);     // A[dd][s]
#pragma unroll
        for (int f0 = 0; f0 < 64; f0 += 16) {
            v16bf bb = load_b_cont(Rvb + (size_t)t * T_ * C_, C_, f0);
            v8f acc = wmma_bf16(a, bb, vzero());
#pragma unroll
            for (int r = 0; r < 8; ++r) {
                int dd = r + kb8;
                float* p = gacc + baseB + (size_t)(d0 + dd) * T_ * C_ +
                           (size_t)t * C_ + hc + f0 + Ml;
                *p += acc[r];
            }
        }
    }
    __threadfence();
    __syncthreads();

    // ---- phase 6: convert this block's slice to bf16 for the out-projection ----
    for (int i = threadIdx.x; i < 16 * 32 * 64; i += 128) {
        int f = i & 63;
        int r = i >> 6;
        int t = r & 31;
        int dd = r >> 5;
        size_t idx = baseB + (size_t)(d0 + dd) * T_ * C_ + (size_t)t * C_ + hc + f;
        aout[idx] = (bf16)gacc[idx];
    }
#undef SA
}

// ---------------- host side ----------------
static void launch_gemm(const bf16* A, const bf16* Wt, const float* bias,
                        const float* resid, float* outf, bf16* outh,
                        int M, int N, int K, hipStream_t s) {
    dim3 g(M / 64, N / 64);
    gemm_kernel<<<g, 128, 0, s>>>(A, Wt, bias, resid, outf, outh, M, N, K);
}

extern "C" void kernel_launch(void* const* d_in, const int* in_sizes, int n_in,
                              void* d_out, int out_size, void* d_ws, size_t ws_size,
                              hipStream_t stream) {
    const float* x    = (const float*)d_in[0];
    const float* temb = (const float*)d_in[1];
    const int*   fi   = (const int*)d_in[2];
    const float* ln_w = (const float*)d_in[3];
    const float* ln_b = (const float*)d_in[4];
    const float* wq = (const float*)d_in[5];
    const float* wk = (const float*)d_in[6];
    const float* wv = (const float*)d_in[7];
    const float* wo = (const float*)d_in[8];
    const float* bo = (const float*)d_in[9];
    const float* dist_w[3] = { (const float*)d_in[10], (const float*)d_in[16], (const float*)d_in[22] };
    const float* dist_b[3] = { (const float*)d_in[11], (const float*)d_in[17], (const float*)d_in[23] };
    const float* time_w[3] = { (const float*)d_in[12], (const float*)d_in[18], (const float*)d_in[24] };
    const float* time_b[3] = { (const float*)d_in[13], (const float*)d_in[19], (const float*)d_in[25] };
    const float* out_w[3]  = { (const float*)d_in[14], (const float*)d_in[20], (const float*)d_in[26] };
    const float* out_b[3]  = { (const float*)d_in[15], (const float*)d_in[21], (const float*)d_in[27] };

    char* ws = (char*)d_ws;
    size_t off = 0;
    auto alloc = [&](size_t bytes) -> void* {
        void* p = ws + off;
        off += (bytes + 255) & ~(size_t)255;
        return p;
    };
    const size_t SZ = (size_t)MROWS * C_;
    bf16* xn   = (bf16*)alloc(SZ * 2);
    bf16* qb   = (bf16*)alloc(SZ * 2);
    bf16* kb   = (bf16*)alloc(SZ * 2);
    bf16* vb   = (bf16*)alloc(SZ * 2);
    bf16* aout = (bf16*)alloc(SZ * 2);
    float* gacc = (float*)alloc(SZ * 4);
    bf16* wqt = (bf16*)alloc((size_t)C_ * C_ * 2);
    bf16* wkt = (bf16*)alloc((size_t)C_ * C_ * 2);
    bf16* wvt = (bf16*)alloc((size_t)C_ * C_ * 2);
    bf16* wot = (bf16*)alloc((size_t)C_ * C_ * 2);
    bf16* owt[3], *twt[3], *R[3];
    for (int i = 0; i < 3; ++i) owt[i] = (bf16*)alloc((size_t)C_ * C_ * 2);
    for (int i = 0; i < 3; ++i) twt[i] = (bf16*)alloc((size_t)C_ * TE_ * 2);
    bf16* tembh = (bf16*)alloc((size_t)B_ * T_ * TE_ * 2);
    float* te   = (float*)alloc((size_t)B_ * T_ * C_ * 4);
    bf16* embA  = (bf16*)alloc((size_t)B_ * T_ * T_ * C_ * 2);
    for (int i = 0; i < 3; ++i) R[i] = (bf16*)alloc((size_t)B_ * T_ * T_ * C_ * 2);

    // 1) weight transpose+convert
    int sqBlocks = (C_ * C_ + 255) / 256;
    tcvt_kernel<<<sqBlocks, 256, 0, stream>>>(wq, wqt, C_, C_);
    tcvt_kernel<<<sqBlocks, 256, 0, stream>>>(wk, wkt, C_, C_);
    tcvt_kernel<<<sqBlocks, 256, 0, stream>>>(wv, wvt, C_, C_);
    tcvt_kernel<<<sqBlocks, 256, 0, stream>>>(wo, wot, C_, C_);
    for (int i = 0; i < 3; ++i) {
        tcvt_kernel<<<sqBlocks, 256, 0, stream>>>(out_w[i], owt[i], C_, C_);
        tcvt_kernel<<<(C_ * TE_ + 255) / 256, 256, 0, stream>>>(time_w[i], twt[i], C_, TE_);
    }
    // 2) temb -> bf16
    long ntemb = (long)B_ * T_ * TE_;
    cvt_kernel<<<(int)((ntemb + 255) / 256), 256, 0, stream>>>(temb, tembh, ntemb);
    // 3) LayerNorm
    ln_kernel<<<(int)(MROWS / 8), 256, 0, stream>>>(x, ln_w, ln_b, xn);
    // 4) QKV projections (WMMA, async-staged B tiles)
    launch_gemm(xn, wqt, nullptr, nullptr, nullptr, qb, (int)MROWS, C_, C_, stream);
    launch_gemm(xn, wkt, nullptr, nullptr, nullptr, kb, (int)MROWS, C_, C_, stream);
    launch_gemm(xn, wvt, nullptr, nullptr, nullptr, vb, (int)MROWS, C_, C_, stream);
    // 5) RPE tensors: te = temb@tw^T+tb ; emb = silu(te + de@dw^T+db) ; R = emb@ow^T+ob
    long nemb = (long)B_ * T_ * T_ * C_;
    for (int i = 0; i < 3; ++i) {
        launch_gemm(tembh, twt[i], time_b[i], nullptr, te, nullptr, B_ * T_, C_, TE_, stream);
        rpe_silu_kernel<<<(int)((nemb + 255) / 256), 256, 0, stream>>>(te, dist_w[i], dist_b[i], fi, embA);
        launch_gemm(embA, owt[i], out_b[i], nullptr, nullptr, R[i], B_ * T_ * T_, C_, C_, stream);
    }
    // 6) fused attention core (WMMA + wave-private async staging)
    size_t attnLds = 16 * 32 * 32 * 4 + 4 * STG_BYTES;   // 64KB + 18KB
    attn_kernel<<<B_ * H_ * (D_ / 16), 128, attnLds, stream>>>(qb, kb, vb, R[0], R[1], R[2], gacc, aout);
    // 7) out-projection + bias + residual -> d_out (f32)
    launch_gemm(aout, wot, bo, x, (float*)d_out, nullptr, (int)MROWS, C_, C_, stream);
    (void)in_sizes; (void)n_in; (void)out_size; (void)ws_size;
}
